// Self_Attn_34986803593303
// MI455X (gfx1250) — matrix operand order repaired and checked by reading, not compiled, not run on hardware
//
#include <hip/hip_runtime.h>

#define B_ 32
#define C_ 512
#define N_ 1024
#define TK 32

typedef __attribute__((ext_vector_type(16))) __bf16 v16bf;
typedef __attribute__((ext_vector_type(8)))  float  v8f;
typedef __attribute__((ext_vector_type(4)))  unsigned int v4u;
typedef __attribute__((ext_vector_type(8)))  int v8i_t;
typedef __attribute__((ext_vector_type(4)))  int v4i_t;

union FragU { v16bf v; unsigned int u[8]; };

__device__ __forceinline__ unsigned short f2bf(float f) {
    unsigned int u = __float_as_uint(f);
    u += 0x7FFFu + ((u >> 16) & 1u);          // round-to-nearest-even
    return (unsigned short)(u >> 16);
}

__device__ __forceinline__ unsigned long long pk4(float4 f) {
    return (unsigned long long)f2bf(f.x)
         | ((unsigned long long)f2bf(f.y) << 16)
         | ((unsigned long long)f2bf(f.z) << 32)
         | ((unsigned long long)f2bf(f.w) << 48);
}

// ---- Tensor Data Mover: 2D bf16 tile (tileW x tileH elems) global -> LDS ----
__device__ __forceinline__ void tdm_load_2d(unsigned int ldsOff,
                                            const unsigned short* gptr,
                                            unsigned int rowLen,   // tensor_dim0
                                            unsigned int nRows,    // tensor_dim1
                                            unsigned int tileW,    // tile_dim0
                                            unsigned int tileH)    // tile_dim1
{
    unsigned long long ga = (unsigned long long)(size_t)gptr;
    v4u g0;
    g0.x = 1u;                                                    // count=1
    g0.y = ldsOff;                                                // lds_addr
    g0.z = (unsigned int)ga;                                      // global lo
    g0.w = ((unsigned int)(ga >> 32) & 0x01FFFFFFu) | (2u << 30); // hi | type=2

    union { unsigned long long q[4]; v8i_t v; } g1;
    g1.q[0] = (1ull << 16)                                        // data_size=2B
            | ((unsigned long long)(rowLen & 0xFFFFu) << 48);     // dim0 lo
    g1.q[1] = (unsigned long long)((rowLen >> 16) & 0xFFFFu)      // dim0 hi
            | ((unsigned long long)nRows << 16)                   // tensor_dim1
            | ((unsigned long long)(tileW & 0xFFFFu) << 48);      // tile_dim0
    g1.q[2] = (unsigned long long)(tileH & 0xFFFFu)               // tile_dim1
            | ((unsigned long long)rowLen << 32);                 // dim0 stride
    g1.q[3] = 0ull;

    v4i_t z4 = {0, 0, 0, 0};
#if __has_include(<hip/amd_detail/amd_gfx1250_TDM.h>)
    v8i_t z8 = {0, 0, 0, 0, 0, 0, 0, 0};
    __builtin_amdgcn_tensor_load_to_lds(g0, g1.v, z4, z4, z8, 0);
#else
    __builtin_amdgcn_tensor_load_to_lds(g0, g1.v, z4, z4, 0);
#endif
}

// A-matrix 16x32 bf16 fragment. LDS layout: ls[M][K] (K contiguous, TK=32).
__device__ __forceinline__ v16bf frag_a(const unsigned short* ls, int mBase, int lane) {
    FragU f;
    const unsigned short* p = ls + (mBase + (lane & 15)) * TK + ((lane >> 4) << 3);
    f.u[0] = *(const unsigned int*)(p + 0);
    f.u[1] = *(const unsigned int*)(p + 2);
    f.u[2] = *(const unsigned int*)(p + 4);
    f.u[3] = *(const unsigned int*)(p + 6);
    f.u[4] = *(const unsigned int*)(p + 16);
    f.u[5] = *(const unsigned int*)(p + 18);
    f.u[6] = *(const unsigned int*)(p + 20);
    f.u[7] = *(const unsigned int*)(p + 22);
    return f.v;
}

// B-matrix 32x16 bf16 fragment. LDS layout: ls[N][K] (K contiguous).
__device__ __forceinline__ v16bf frag_b(const unsigned short* ls, int nBase, int lane) {
    FragU f;
    const unsigned short* p = ls + (nBase + (lane & 15)) * TK + ((lane >> 4) << 3);
#pragma unroll
    for (int i = 0; i < 8; ++i) f.u[i] = *(const unsigned int*)(p + 2 * (i & 3) + ((i >> 2) << 4));
    return f.v;
}

__device__ __forceinline__ v8f wmma_bf16(v16bf a, v16bf b, v8f c) {
    return __builtin_amdgcn_wmma_f32_16x16x32_bf16(false, a, false, b, (short)0, c,
                                                   false, false);
}

// Stage 128x32 f32 W tile -> bf16 LDS [M][K]; phase-split for MLP.
__device__ __forceinline__ void stage_w(unsigned short* dst, const float* Wm,
                                        int o0, int k0, int t) {
    float4 w4[4];
#pragma unroll
    for (int i = 0; i < 4; ++i) {
        int idx = t + i * 256;
        int r = idx >> 3, cq = (idx & 7) << 2;
        w4[i] = *(const float4*)(Wm + (size_t)(o0 + r) * C_ + k0 + cq);
    }
#pragma unroll
    for (int i = 0; i < 4; ++i) {
        int idx = t + i * 256;
        int r = idx >> 3, cq = (idx & 7) << 2;
        *(unsigned long long*)(dst + r * TK + cq) = pk4(w4[i]);
    }
}

// Stage 128x32 f32 attention tile -> bf16 LDS [N][K]; phase-split.
__device__ __forceinline__ void stage_att(unsigned short* dst, const float* A,
                                          int n0, int k0, int t) {
    float4 f4[4];
#pragma unroll
    for (int i = 0; i < 4; ++i) {
        int idx = t + i * 256;
        int nn = idx >> 3, mq = (idx & 7) << 2;
        f4[i] = *(const float4*)(A + (size_t)(n0 + nn) * N_ + k0 + mq);
    }
#pragma unroll
    for (int i = 0; i < 4; ++i) {
        int idx = t + i * 256;
        int nn = idx >> 3, mq = (idx & 7) << 2;
        *(unsigned long long*)(dst + nn * TK + mq) = pk4(f4[i]);
    }
}

// ---------- Kernel 0: xT[b][n][c] = bf16(x[b][c][n])  (LDS-tiled transpose) --
__global__ __launch_bounds__(256) void xt_kernel(const float* __restrict__ x,
                                                 unsigned short* __restrict__ xT) {
    __shared__ unsigned short tile[64 * 68];
    const int n0 = blockIdx.x * 64, c0 = blockIdx.y * 64, b = blockIdx.z;
    const float* X = x + (size_t)b * C_ * N_;
    unsigned short* XT = xT + (size_t)b * N_ * C_;
    const int t = threadIdx.x;

    float4 f4[4];
#pragma unroll
    for (int i = 0; i < 4; ++i) {
        int idx = t + i * 256;
        int cc = idx >> 4, nq = (idx & 15) << 2;        // read coalesced along n
        f4[i] = *(const float4*)(X + (size_t)(c0 + cc) * N_ + n0 + nq);
    }
#pragma unroll
    for (int i = 0; i < 4; ++i) {
        int idx = t + i * 256;
        int cc = idx >> 4, nq = (idx & 15) << 2;
        tile[(nq + 0) * 68 + cc] = f2bf(f4[i].x);
        tile[(nq + 1) * 68 + cc] = f2bf(f4[i].y);
        tile[(nq + 2) * 68 + cc] = f2bf(f4[i].z);
        tile[(nq + 3) * 68 + cc] = f2bf(f4[i].w);
    }
    __syncthreads();
#pragma unroll
    for (int i = 0; i < 4; ++i) {
        int idx = t + i * 256;
        int nn = idx >> 4, cq = (idx & 15) << 2;        // write coalesced along c
        *(unsigned long long*)(XT + (size_t)(n0 + nn) * C_ + c0 + cq) =
            *(const unsigned long long*)(tile + nn * 68 + cq);
    }
}

// ---------- Kernel 1: QKV projections (3 GEMMs via blockIdx.z) ---------------
// Q,K written transposed bf16 [b][n][c]; V written bf16 [b][c][n].
__global__ __launch_bounds__(256) void qkv_proj_kernel(
    const unsigned short* __restrict__ xT,
    const float* __restrict__ Wq, const float* __restrict__ bq,
    const float* __restrict__ Wk, const float* __restrict__ bk,
    const float* __restrict__ Wv, const float* __restrict__ bv,
    unsigned short* __restrict__ qt,
    unsigned short* __restrict__ kT,
    unsigned short* __restrict__ vbf)
{
    __shared__ unsigned short lsA[2][128 * TK];
    __shared__ unsigned short lsB[2][128 * TK];

    const int n0    = blockIdx.x * 128;
    const int o0    = blockIdx.y * 128;
    const int which = blockIdx.z / B_;
    const int b     = blockIdx.z % B_;

    const float* Wm   = (which == 0) ? Wq : (which == 1) ? Wk : Wv;
    const float* bias = (which == 0) ? bq : (which == 1) ? bk : bv;
    const unsigned short* Xb = xT + (size_t)b * N_ * C_ + (size_t)n0 * C_;

    const int t = threadIdx.x, lane = t & 31, wave = t >> 5;
    const int wm = wave & 3, wn = wave >> 2;
    const unsigned int bOff = (unsigned int)(size_t)(void*)&lsB[0][0];

    if (wave == 0) tdm_load_2d(bOff, Xb, C_, N_, TK, 128);   // B tile 0
    stage_w(lsA[0], Wm, o0, 0, t);                           // A tile 0

    v8f acc[2][4] = {};
    int buf = 0;
    for (int k0 = 0; k0 < C_; k0 += TK, buf ^= 1) {
        if (wave == 0) __builtin_amdgcn_s_wait_tensorcnt(0);
        __syncthreads();
        const int k1 = k0 + TK;
        if (k1 < C_) {                                       // prefetch next slab
            if (wave == 0) tdm_load_2d(bOff + ((buf ^ 1) << 13), Xb + k1,
                                       C_, N_, TK, 128);
            stage_w(lsA[buf ^ 1], Wm, o0, k1, t);
        }
        const unsigned short* cA = lsA[buf];
        const unsigned short* cB = lsB[buf];
        v16bf af[2], bfr[4];
#pragma unroll
        for (int mi = 0; mi < 2; ++mi) af[mi] = frag_a(cA, wm * 32 + mi * 16, lane);
#pragma unroll
        for (int ni = 0; ni < 4; ++ni) bfr[ni] = frag_b(cB, wn * 64 + ni * 16, lane);
#pragma unroll
        for (int mi = 0; mi < 2; ++mi)
#pragma unroll
            for (int ni = 0; ni < 4; ++ni)
                acc[mi][ni] = wmma_bf16(af[mi], bfr[ni], acc[mi][ni]);
    }

    unsigned short* dstT = (which == 0) ? qt : kT;           // [b][n][c] layouts
    const int laneN = lane & 15, rowAdd = (lane >> 4) << 3;
#pragma unroll
    for (int mi = 0; mi < 2; ++mi)
#pragma unroll
        for (int r = 0; r < 8; ++r) {
            int oo = o0 + wm * 32 + mi * 16 + r + rowAdd;
            float bb = bias[oo];
#pragma unroll
            for (int ni = 0; ni < 4; ++ni) {
                int nn = n0 + wn * 64 + ni * 16 + laneN;
                unsigned short h = f2bf(acc[mi][ni][r] + bb);
                if (which < 2)
                    dstT[(size_t)b * N_ * C_ + (size_t)nn * C_ + oo] = h;
                else
                    vbf[(size_t)b * C_ * N_ + (size_t)oo * N_ + nn] = h;
            }
        }
}

// ---------- Kernel 2: energy[b][n][m] = sum_c Qt[n][c]*kT[m][c] --------------
__global__ __launch_bounds__(256) void energy_kernel(
    const unsigned short* __restrict__ qt,
    const unsigned short* __restrict__ kT,
    float* __restrict__ att)
{
    __shared__ unsigned short lsA[2][128 * TK];
    __shared__ unsigned short lsB[2][128 * TK];

    const int m0  = blockIdx.x * 128;
    const int nr0 = blockIdx.y * 128;
    const int b   = blockIdx.z;
    const unsigned short* Qb = qt + (size_t)b * N_ * C_ + (size_t)nr0 * C_;
    const unsigned short* Kb = kT + (size_t)b * N_ * C_ + (size_t)m0 * C_;

    const int t = threadIdx.x, lane = t & 31, wave = t >> 5;
    const int wm = wave & 3, wn = wave >> 2;
    const unsigned int aOff = (unsigned int)(size_t)(void*)&lsA[0][0];
    const unsigned int bOff = (unsigned int)(size_t)(void*)&lsB[0][0];

    if (wave == 0) {                                         // tiles 0 via TDM
        tdm_load_2d(aOff, Qb, C_, N_, TK, 128);
        tdm_load_2d(bOff, Kb, C_, N_, TK, 128);
    }

    v8f acc[2][4] = {};
    int buf = 0;
    for (int k0 = 0; k0 < C_; k0 += TK, buf ^= 1) {
        if (wave == 0) __builtin_amdgcn_s_wait_tensorcnt(0);
        __syncthreads();
        const int k1 = k0 + TK;
        if (k1 < C_ && wave == 0) {                          // prefetch next slab
            tdm_load_2d(aOff + ((buf ^ 1) << 13), Qb + k1, C_, N_, TK, 128);
            tdm_load_2d(bOff + ((buf ^ 1) << 13), Kb + k1, C_, N_, TK, 128);
        }
        const unsigned short* cA = lsA[buf];
        const unsigned short* cB = lsB[buf];
        v16bf af[2], bfr[4];
#pragma unroll
        for (int mi = 0; mi < 2; ++mi) af[mi] = frag_a(cA, wm * 32 + mi * 16, lane);
#pragma unroll
        for (int ni = 0; ni < 4; ++ni) bfr[ni] = frag_b(cB, wn * 64 + ni * 16, lane);
#pragma unroll
        for (int mi = 0; mi < 2; ++mi)
#pragma unroll
            for (int ni = 0; ni < 4; ++ni)
                acc[mi][ni] = wmma_bf16(af[mi], bfr[ni], acc[mi][ni]);
    }

    const int laneN = lane & 15, rowAdd = (lane >> 4) << 3;
#pragma unroll
    for (int mi = 0; mi < 2; ++mi)
#pragma unroll
        for (int r = 0; r < 8; ++r) {
            int nn = nr0 + wm * 32 + mi * 16 + r + rowAdd;
#pragma unroll
            for (int ni = 0; ni < 4; ++ni) {
                int mm = m0 + wn * 64 + ni * 16 + laneN;
                att[(size_t)b * N_ * N_ + (size_t)nn * N_ + mm] = acc[mi][ni][r];
            }
        }
}

// ---------- Kernel 3: row softmax (one wave32 per row of 1024) ---------------
__global__ __launch_bounds__(256) void softmax_kernel(float* __restrict__ att) {
    const int lane = threadIdx.x & 31;
    const int wave = threadIdx.x >> 5;
    const int row  = blockIdx.x * 8 + wave;
    float* p = att + (size_t)row * N_;

    float vals[32];
    float m = -3.402823466e+38f;
#pragma unroll
    for (int j = 0; j < 32; ++j) { vals[j] = p[lane + j * 32]; m = fmaxf(m, vals[j]); }
#pragma unroll
    for (int off = 16; off; off >>= 1) m = fmaxf(m, __shfl_xor(m, off, 32));
    float s = 0.f;
#pragma unroll
    for (int j = 0; j < 32; ++j) { vals[j] = __expf(vals[j] - m); s += vals[j]; }
#pragma unroll
    for (int off = 16; off; off >>= 1) s += __shfl_xor(s, off, 32);
    float inv = 1.0f / s;
#pragma unroll
    for (int j = 0; j < 32; ++j) p[lane + j * 32] = vals[j] * inv;
}

// ---------- Kernel 4: out[b][c][n] = gamma*sum_m V[c][m]*A[n][m] + x ---------
__global__ __launch_bounds__(256) void out_kernel(
    const unsigned short* __restrict__ vbf,
    const float* __restrict__ att,
    const float* __restrict__ x,
    const float* __restrict__ gamma,
    float* __restrict__ outp)
{
    __shared__ unsigned short lsA[2][128 * TK];
    __shared__ unsigned short lsB[2][128 * TK];

    const int n0 = blockIdx.x * 128;
    const int c0 = blockIdx.y * 128;
    const int b  = blockIdx.z;
    const unsigned short* Vb = vbf + (size_t)b * C_ * N_ + (size_t)c0 * N_;
    const float* A = att + (size_t)b * N_ * N_;

    const int t = threadIdx.x, lane = t & 31, wave = t >> 5;
    const int wm = wave & 3, wn = wave >> 2;
    const unsigned int aOff = (unsigned int)(size_t)(void*)&lsA[0][0];

    if (wave == 0) tdm_load_2d(aOff, Vb, N_, C_, TK, 128);   // A tile 0
    stage_att(lsB[0], A, n0, 0, t);                          // B tile 0

    v8f acc[2][4] = {};
    int buf = 0;
    for (int k0 = 0; k0 < N_; k0 += TK, buf ^= 1) {          // K-dim = m
        if (wave == 0) __builtin_amdgcn_s_wait_tensorcnt(0);
        __syncthreads();
        const int k1 = k0 + TK;
        if (k1 < N_) {                                       // prefetch next slab
            if (wave == 0) tdm_load_2d(aOff + ((buf ^ 1) << 13), Vb + k1,
                                       N_, C_, TK, 128);
            stage_att(lsB[buf ^ 1], A, n0, k1, t);
        }
        const unsigned short* cA = lsA[buf];
        const unsigned short* cB = lsB[buf];
        v16bf af[2], bfr[4];
#pragma unroll
        for (int mi = 0; mi < 2; ++mi) af[mi] = frag_a(cA, wm * 32 + mi * 16, lane);
#pragma unroll
        for (int ni = 0; ni < 4; ++ni) bfr[ni] = frag_b(cB, wn * 64 + ni * 16, lane);
#pragma unroll
        for (int mi = 0; mi < 2; ++mi)
#pragma unroll
            for (int ni = 0; ni < 4; ++ni)
                acc[mi][ni] = wmma_bf16(af[mi], bfr[ni], acc[mi][ni]);
    }

    const float g = gamma[0];
    const int laneN = lane & 15, rowAdd = (lane >> 4) << 3;
#pragma unroll
    for (int mi = 0; mi < 2; ++mi)
#pragma unroll
        for (int r = 0; r < 8; ++r) {
            int cc = c0 + wm * 32 + mi * 16 + r + rowAdd;
#pragma unroll
            for (int ni = 0; ni < 4; ++ni) {
                int nn = n0 + wn * 64 + ni * 16 + laneN;
                size_t idx = ((size_t)b * C_ + cc) * N_ + nn;
                outp[idx] = g * acc[mi][ni][r] + x[idx];
            }
        }
}

extern "C" void kernel_launch(void* const* d_in, const int* in_sizes, int n_in,
                              void* d_out, int out_size, void* d_ws, size_t ws_size,
                              hipStream_t stream) {
    const float* x     = (const float*)d_in[0];
    const float* Wq    = (const float*)d_in[1];
    const float* bq    = (const float*)d_in[2];
    const float* Wk    = (const float*)d_in[3];
    const float* bk    = (const float*)d_in[4];
    const float* Wv    = (const float*)d_in[5];
    const float* bv    = (const float*)d_in[6];
    const float* gamma = (const float*)d_in[7];

    float* outp = (float*)d_out;                         // [B,C,W,H]
    float* att  = outp + (size_t)B_ * C_ * N_;           // [B,N,N]

    unsigned short* qt  = (unsigned short*)d_ws;                 // [B][N][C] bf16
    unsigned short* kT  = qt  + (size_t)B_ * N_ * C_;            // [B][N][C] bf16
    unsigned short* vbf = kT  + (size_t)B_ * N_ * C_;            // [B][C][N] bf16
    unsigned short* xT  = vbf + (size_t)B_ * C_ * N_;            // [B][N][C] bf16

    dim3 blk(256);
    xt_kernel<<<dim3(N_ / 64, C_ / 64, B_), blk, 0, stream>>>(x, xT);
    qkv_proj_kernel<<<dim3(N_ / 128, C_ / 128, 3 * B_), blk, 0, stream>>>(
        xT, Wq, bq, Wk, bk, Wv, bv, qt, kT, vbf);
    energy_kernel<<<dim3(N_ / 128, N_ / 128, B_), blk, 0, stream>>>(qt, kT, att);
    softmax_kernel<<<dim3(B_ * N_ / 8), blk, 0, stream>>>(att);
    out_kernel<<<dim3(N_ / 128, C_ / 128, B_), blk, 0, stream>>>(vbf, att, x, gamma, outp);
}
